// MHA_59854664237467
// MI455X (gfx1250) — matrix operand order repaired and checked by reading, not compiled, not run on hardware
//
#include <hip/hip_runtime.h>

#define NB 16
#define CC 384
#define NN 1024
#define NHD 8
#define HD 48
#define HDP 64
#define ATTN_SCALE 0.14433756729740643f  // 1/sqrt(48)

typedef __attribute__((ext_vector_type(16))) __bf16 v16bf;
typedef __attribute__((ext_vector_type(8)))  float  v8f;

union FragB { v16bf v; uint4 q[2]; };

__device__ inline unsigned short f2bf(float f) {
  unsigned u = __float_as_uint(f);
  u += 0x7FFFu + ((u >> 16) & 1u);   // round-to-nearest-even
  return (unsigned short)(u >> 16);
}

// gfx1250 async global->LDS copy, GV mode (64-bit vaddr, saddr=off).
// VDST = LDS byte address (low 32 bits of flat shared pointer), tracked by ASYNCcnt.
__device__ __forceinline__ void async_ld_b128(unsigned ldsOff, const void* g) {
  asm volatile("global_load_async_to_lds_b128 %0, %1, off"
               :: "v"(ldsOff), "v"(g) : "memory");
}
__device__ __forceinline__ void wait_async_le2() {
  asm volatile("s_wait_asynccnt 2" ::: "memory");
}
__device__ __forceinline__ void wait_async_0() {
  asm volatile("s_wait_asynccnt 0" ::: "memory");
}

// ---------------------------------------------------------------- utilities
__global__ void k_zero(uint4* __restrict__ p, long n4) {
  size_t i = (size_t)blockIdx.x * blockDim.x + threadIdx.x;
  size_t stride = (size_t)gridDim.x * blockDim.x;
  uint4 z = make_uint4(0u, 0u, 0u, 0u);
  for (; i < (size_t)n4; i += stride) p[i] = z;
}

__global__ void k_cvt_w(const float* __restrict__ wq, const float* __restrict__ wp,
                        unsigned short* __restrict__ wqb, unsigned short* __restrict__ wpb) {
  int i = blockIdx.x * 256 + threadIdx.x;
  const int nq = 3 * CC * CC;
  const int np = CC * CC;
  if (i < nq) wqb[i] = f2bf(wq[i]);
  if (i < np) wpb[i] = f2bf(wp[i]);
}

// x: (B, C, N) f32  ->  xbf: (B, N, C) bf16   (write-coalesced transpose)
__global__ void k_cvt_x(const float* __restrict__ x, unsigned short* __restrict__ xbf) {
  int i = blockIdx.x * 256 + threadIdx.x;
  if (i >= NB * NN * CC) return;
  int c = i % CC;
  int n = (i / CC) % NN;
  int b = i / (CC * NN);
  xbf[i] = f2bf(x[((size_t)b * CC + c) * NN + n]);
}

// ---------------------------------------------------------------- QKV GEMM
__global__ __launch_bounds__(32) void k_qkv(const unsigned short* __restrict__ xbf,
                                            const unsigned short* __restrict__ wqb,
                                            unsigned short* __restrict__ Qp,
                                            unsigned short* __restrict__ Kp,
                                            unsigned short* __restrict__ Vt) {
  int tileM = blockIdx.x, tileO = blockIdx.y, b = blockIdx.z;
  int lane = threadIdx.x;
  int lo15 = lane & 15;
  bool hi = lane >= 16;
  const unsigned short* A = xbf + ((size_t)b * NN + tileM * 16 + lo15) * CC;
  const unsigned short* W = wqb + (size_t)(tileO * 16 + lo15) * CC;
  v8f acc = {};
#pragma unroll
  for (int k0 = 0; k0 < CC; k0 += 32) {
    FragB fa, fb;
    const unsigned short* pa = A + k0 + (hi ? 16 : 0);  // A: K 0..15 / 16..31 contiguous
    fa.q[0] = *(const uint4*)(pa);
    fa.q[1] = *(const uint4*)(pa + 8);
    const unsigned short* pw = W + k0 + (hi ? 16 : 0);  // B: K 0..15 / 16..31 contiguous
    fb.q[0] = *(const uint4*)(pw);
    fb.q[1] = *(const uint4*)(pw + 8);
    acc = __builtin_amdgcn_wmma_f32_16x16x32_bf16(false, fa.v, false, fb.v,
                                                  (short)0, acc, false, false);
  }
  int o0 = tileO * 16;                 // 16-wide tiles never straddle a head
  int s = o0 / CC, rem = o0 % CC;
  int h = rem / HD;
  int d = (rem % HD) + lo15;
  int mh = hi ? 8 : 0;
  size_t bh = (size_t)(b * NHD + h);
#pragma unroll
  for (int r = 0; r < 8; ++r) {
    int n = tileM * 16 + r + mh;
    unsigned short v = f2bf(acc[r]);
    if (s == 0)      Qp[(bh * NN + n) * HDP + d] = v;
    else if (s == 1) Kp[(bh * NN + n) * HDP + d] = v;
    else             Vt[(bh * HDP + d) * NN + n] = v;
  }
}

// ---------------------------------------------------------------- attention
// 8 waves / workgroup; each wave owns 16 query rows (128 rows per block).
// K (32x64) and V^T (64x32) chunks are async-copied global->LDS, double
// buffered (every wave issues exactly 2 async instructions per chunk so
// s_wait_asynccnt 2 pipelines one chunk ahead uniformly).
__global__ __launch_bounds__(256) void k_attn(const unsigned short* __restrict__ Qp,
                                              const unsigned short* __restrict__ Kp,
                                              const unsigned short* __restrict__ Vt,
                                              unsigned short* __restrict__ Obuf) {
  __shared__ unsigned short sK[2][32][HDP];   // 2 x 4KB
  __shared__ unsigned short sV[2][HDP][32];   // 2 x 4KB (incl. zero pad rows 48..63)
  __shared__ unsigned short sP[8][16][32];    // per-wave P staging, 8KB

  int bh = blockIdx.y;
  int b = bh / NHD, h = bh % NHD;
  int tid = threadIdx.x;
  int w = tid >> 5;
  int lane = tid & 31;
  int lo15 = lane & 15;
  bool hi = lane >= 16;
  int mh = hi ? 8 : 0;
  int ab = hi ? 16 : 0;
  int m0 = blockIdx.x * 128 + w * 16;

  const unsigned short* Kb = Kp + (size_t)bh * NN * HDP;
  const unsigned short* Vb = Vt + (size_t)bh * HDP * NN;

  // staging assignment: 256 threads x 16B cover each 4KB chunk
  int krow = tid >> 3, kcg = (tid & 7) * 8;   // K: 32 rows x 64 elems
  int vrow = tid >> 2, vcg = (tid & 3) * 8;   // V: 64 rows x 32 elems
  auto stage = [&](int buf, int n0) {
    async_ld_b128((unsigned)(size_t)&sK[buf][krow][kcg],
                  Kb + (size_t)(n0 + krow) * HDP + kcg);
    async_ld_b128((unsigned)(size_t)&sV[buf][vrow][vcg],
                  Vb + (size_t)vrow * NN + n0 + vcg);
  };

  const unsigned short* Qr = Qp + ((size_t)bh * NN + m0 + lo15) * HDP;
  FragB qa0, qa1;                      // Q 16x64 = two 16x32 A-fragments
  qa0.q[0] = *(const uint4*)(Qr + ab);
  qa0.q[1] = *(const uint4*)(Qr + ab + 8);
  qa1.q[0] = *(const uint4*)(Qr + 32 + ab);
  qa1.q[1] = *(const uint4*)(Qr + 32 + ab + 8);

  float rowmax[8], rowsum[8];
#pragma unroll
  for (int r = 0; r < 8; ++r) { rowmax[r] = -1e30f; rowsum[r] = 0.f; }
  v8f acc0 = {}, acc1 = {}, acc2 = {};

  const int NITER = NN / 32;
  stage(0, 0);
  for (int it = 0; it < NITER; ++it) {
    int buf = it & 1;
    if (it + 1 < NITER) {
      stage(buf ^ 1, (it + 1) * 32);
      wait_async_le2();                // current buf's 2 copies complete (in-order)
    } else {
      wait_async_0();
    }
    __syncthreads();

    // ---- scores: two 16x16 tiles from LDS K chunk
    v8f s0 = {}, s1 = {};
    {
      const unsigned short* Kr0 = &sK[buf][lo15][hi ? 16 : 0];
      const unsigned short* Kr1 = &sK[buf][16 + lo15][hi ? 16 : 0];
      FragB f;
      f.q[0] = *(const uint4*)(Kr0);      f.q[1] = *(const uint4*)(Kr0 + 8);
      s0 = __builtin_amdgcn_wmma_f32_16x16x32_bf16(false, qa0.v, false, f.v, (short)0, s0, false, false);
      f.q[0] = *(const uint4*)(Kr0 + 32); f.q[1] = *(const uint4*)(Kr0 + 40);
      s0 = __builtin_amdgcn_wmma_f32_16x16x32_bf16(false, qa1.v, false, f.v, (short)0, s0, false, false);
      f.q[0] = *(const uint4*)(Kr1);      f.q[1] = *(const uint4*)(Kr1 + 8);
      s1 = __builtin_amdgcn_wmma_f32_16x16x32_bf16(false, qa0.v, false, f.v, (short)0, s1, false, false);
      f.q[0] = *(const uint4*)(Kr1 + 32); f.q[1] = *(const uint4*)(Kr1 + 40);
      s1 = __builtin_amdgcn_wmma_f32_16x16x32_bf16(false, qa1.v, false, f.v, (short)0, s1, false, false);
    }
    // ---- online softmax (rows live in 16-lane half-groups of D layout)
    float corr[8];
#pragma unroll
    for (int r = 0; r < 8; ++r) {
      float a0 = s0[r] * ATTN_SCALE;
      float a1 = s1[r] * ATTN_SCALE;
      float mx = fmaxf(a0, a1);
      mx = fmaxf(mx, __shfl_xor(mx, 1));
      mx = fmaxf(mx, __shfl_xor(mx, 2));
      mx = fmaxf(mx, __shfl_xor(mx, 4));
      mx = fmaxf(mx, __shfl_xor(mx, 8));
      float mnew = fmaxf(rowmax[r], mx);
      float p0 = __expf(a0 - mnew);
      float p1 = __expf(a1 - mnew);
      float c  = __expf(rowmax[r] - mnew);
      rowmax[r] = mnew;
      float ps = p0 + p1;
      ps += __shfl_xor(ps, 1);
      ps += __shfl_xor(ps, 2);
      ps += __shfl_xor(ps, 4);
      ps += __shfl_xor(ps, 8);
      rowsum[r] = rowsum[r] * c + ps;
      corr[r] = c;
      int m = r + mh;
      sP[w][m][lo15]      = f2bf(p0);
      sP[w][m][16 + lo15] = f2bf(p1);
    }
#pragma unroll
    for (int r = 0; r < 8; ++r) { acc0[r] *= corr[r]; acc1[r] *= corr[r]; acc2[r] *= corr[r]; }

    // ---- P (16x32) A-fragment from LDS; same-wave DS ops are in-order
    FragB pa;
    pa.q[0] = *(const uint4*)(&sP[w][lo15][ab]);
    pa.q[1] = *(const uint4*)(&sP[w][lo15][ab + 8]);
#pragma unroll
    for (int dt = 0; dt < 3; ++dt) {
      FragB vf;
      const unsigned short* Vr = &sV[buf][dt * 16 + lo15][hi ? 16 : 0];
      vf.q[0] = *(const uint4*)(Vr);
      vf.q[1] = *(const uint4*)(Vr + 8);
      v8f& a = dt == 0 ? acc0 : (dt == 1 ? acc1 : acc2);
      a = __builtin_amdgcn_wmma_f32_16x16x32_bf16(false, pa.v, false, vf.v, (short)0, a, false, false);
    }
    __syncthreads();                   // buf free for the copy 2 iterations ahead
  }
#pragma unroll
  for (int r = 0; r < 8; ++r) {
    int m = r + mh;
    float inv = 1.0f / rowsum[r];
    size_t base = ((size_t)b * NN + m0 + m) * CC + h * HD;
    Obuf[base + lo15]      = f2bf(acc0[r] * inv);
    Obuf[base + 16 + lo15] = f2bf(acc1[r] * inv);
    Obuf[base + 32 + lo15] = f2bf(acc2[r] * inv);
  }
}

// ---------------------------------------------------------------- projection
__global__ __launch_bounds__(32) void k_proj(const unsigned short* __restrict__ Obuf,
                                             const unsigned short* __restrict__ wpb,
                                             const float* __restrict__ bias,
                                             float* __restrict__ out) {
  int tileM = blockIdx.x, tileO = blockIdx.y, b = blockIdx.z;
  int lane = threadIdx.x;
  int lo15 = lane & 15;
  bool hi = lane >= 16;
  const unsigned short* A = Obuf + ((size_t)b * NN + tileM * 16 + lo15) * CC;
  const unsigned short* W = wpb + (size_t)(tileO * 16 + lo15) * CC;
  v8f acc = {};
#pragma unroll
  for (int k0 = 0; k0 < CC; k0 += 32) {
    FragB fa, fb;
    const unsigned short* pa = A + k0 + (hi ? 16 : 0);
    fa.q[0] = *(const uint4*)(pa);
    fa.q[1] = *(const uint4*)(pa + 8);
    const unsigned short* pw = W + k0 + (hi ? 16 : 0);
    fb.q[0] = *(const uint4*)(pw);
    fb.q[1] = *(const uint4*)(pw + 8);
    acc = __builtin_amdgcn_wmma_f32_16x16x32_bf16(false, fa.v, false, fb.v,
                                                  (short)0, acc, false, false);
  }
  int o = tileO * 16 + lo15;
  float bv = bias[o];
  int mh = hi ? 8 : 0;
#pragma unroll
  for (int r = 0; r < 8; ++r) {
    int n = tileM * 16 + r + mh;
    out[((size_t)b * CC + o) * NN + n] = acc[r] + bv;  // (B, C, H*W)
  }
}

// ---------------------------------------------------------------- launcher
extern "C" void kernel_launch(void* const* d_in, const int* in_sizes, int n_in,
                              void* d_out, int out_size, void* d_ws, size_t ws_size,
                              hipStream_t stream) {
  (void)in_sizes; (void)n_in; (void)out_size; (void)ws_size;
  const float* x     = (const float*)d_in[0];
  const float* wqkv  = (const float*)d_in[1];
  const float* wproj = (const float*)d_in[2];
  const float* bproj = (const float*)d_in[3];
  float* out = (float*)d_out;

  unsigned short* ws = (unsigned short*)d_ws;
  size_t off = 0;
  unsigned short* xbf = ws + off; off += (size_t)NB * NN * CC;
  unsigned short* wqb = ws + off; off += (size_t)3 * CC * CC;
  unsigned short* wpb = ws + off; off += (size_t)CC * CC;
  unsigned short* Qp  = ws + off; off += (size_t)NB * NHD * NN * HDP;
  unsigned short* Kp  = ws + off; off += (size_t)NB * NHD * NN * HDP;
  unsigned short* Vt  = ws + off; off += (size_t)NB * NHD * HDP * NN;
  unsigned short* Obuf = xbf;  // reuse: xbf dead after k_qkv

  size_t zeroElems = (size_t)3 * NB * NHD * NN * HDP;  // Qp|Kp|Vt contiguous
  k_zero<<<4096, 256, 0, stream>>>((uint4*)Qp, (long)(zeroElems / 8));

  int nq = 3 * CC * CC;
  k_cvt_w<<<(nq + 255) / 256, 256, 0, stream>>>(wqkv, wproj, wqb, wpb);
  int nx = NB * NN * CC;
  k_cvt_x<<<(nx + 255) / 256, 256, 0, stream>>>(x, xbf);

  k_qkv<<<dim3(NN / 16, 3 * CC / 16, NB), 32, 0, stream>>>(xbf, wqb, Qp, Kp, Vt);
  k_attn<<<dim3(NN / 128, NB * NHD), 256, 0, stream>>>(Qp, Kp, Vt, Obuf);
  k_proj<<<dim3(NN / 16, CC / 16, NB), 32, 0, stream>>>(Obuf, wpb, bproj, out);
}
